// MultiHeadDifferenceImageAttention_87634512708303
// MI455X (gfx1250) — matrix operand order repaired and checked by reading, not compiled, not run on hardware
//
#include <hip/hip_runtime.h>
#include <hip/hip_bf16.h>

typedef __attribute__((ext_vector_type(16))) __bf16 v16bf;
typedef __attribute__((ext_vector_type(8)))  __bf16 v8bf;
typedef __attribute__((ext_vector_type(8)))  float  v8f;

namespace {

constexpr int kB = 16, kS = 14, kC = 2048, kH = 4, kDM = 512;
constexpr int kC8 = kC / 8;       // 256
constexpr int kN  = kS * kS;      // 196
constexpr int kNP = 224;          // N padded to multiple of 32 for tail-free K loops
constexpr int kF  = kC * kN;      // 401408
constexpr int kHB = kH * kB;      // 64
constexpr int kKSplit = 64;
constexpr int kKChunk = kF / kKSplit;  // 6272 (exact)

// ---- workspace layout (element counts) ----
constexpr size_t nXT    = (size_t)kHB * kN * kC;    // x^T[hb][n][c] bf16 (input order!)
constexpr size_t nDiffT = (size_t)kB  * kN * kC;    // diff^T[b][n][c] bf16
constexpr size_t nWq    = (size_t)kH * kC8 * kC;    // bf16
constexpr size_t nWk    = nWq;
constexpr size_t nWv    = (size_t)kH * kC * kC;
constexpr size_t nQ     = (size_t)kHB * kN * kC8;   // Q[hb][n][o]
constexpr size_t nKT    = (size_t)kHB * kN * kC8;   // K^T[hb][m][o]
constexpr size_t nE     = (size_t)kHB * kN * kN;    // energy f32
constexpr size_t nAT    = (size_t)kHB * kN * kNP;   // attn[hb][mq][nk] bf16, nk padded w/ 0
constexpr size_t nV     = (size_t)kHB * kC * kNP;   // V[hb][c][nk] bf16, nk padded w/ 0
constexpr size_t nFeats = (size_t)kHB * kF;         // feats[(b*H+h)][c*N+m] bf16

constexpr size_t oXT    = 0;
constexpr size_t oDiffT = oXT    + nXT    * 2;
constexpr size_t oWq    = oDiffT + nDiffT * 2;
constexpr size_t oWk    = oWq    + nWq    * 2;
constexpr size_t oWv    = oWk    + nWk    * 2;
constexpr size_t oQ     = oWv    + nWv    * 2;
constexpr size_t oKT    = oQ     + nQ     * 2;
constexpr size_t oE     = oKT    + nKT    * 2;
constexpr size_t oAT    = oE     + nE     * 4;
constexpr size_t oV     = oAT    + nAT    * 2;
constexpr size_t oFeats = oV     + nV     * 2;

#define WMMA_BF16(a, b, c) \
  __builtin_amdgcn_wmma_f32_16x16x32_bf16(false, (a), false, (b), (short)0, (c), false, false)

// ---- contiguous fragment loaders (wave32 layouts per CDNA5 ISA 7.12.2) ----

// A 16x32 fragment from row-major [row][k] source. Per lane: 2x 16B loads.
__device__ __forceinline__ v16bf load_a16(const __bf16* __restrict__ A, int ld,
                                          int m0, int k0, int maxRow) {
  int l = (int)(threadIdx.x & 31u);
  int m = m0 + (l & 15); m = (m > maxRow) ? maxRow : m;
  int kh = (l >> 4) * 8;
  const __bf16* p = A + (size_t)m * (size_t)ld + (size_t)(k0 + kh);
  v8bf lo = *(const v8bf*)p;
  v8bf hi = *(const v8bf*)(p + 16);
  return __builtin_shufflevector(lo, hi, 0, 1, 2, 3, 4, 5, 6, 7,
                                 8, 9, 10, 11, 12, 13, 14, 15);
}

// A fragment converted inline from fp32 row-major source (Wl).
__device__ __forceinline__ v16bf load_a_f32(const float* __restrict__ A, int ld,
                                            int m0, int k0) {
  int l = (int)(threadIdx.x & 31u);
  int m = m0 + (l & 15);
  int kh = (l >> 4) * 8;
  const float* p = A + (size_t)m * (size_t)ld + (size_t)(k0 + kh);
  v8f lo = *(const v8f*)p;
  v8f hi = *(const v8f*)(p + 16);
  v16bf r;
#pragma unroll
  for (int j = 0; j < 8; ++j) { r[j] = (__bf16)lo[j]; r[8 + j] = (__bf16)hi[j]; }
  return r;
}

// B 32x16 fragment from [col][k] source (k = 8h+s, 16+8h+s per lane): 2x 16B loads.
__device__ __forceinline__ v16bf load_b16(const __bf16* __restrict__ Bm, int ld,
                                          int k0, int n0, int maxCol) {
  int l = (int)(threadIdx.x & 31u);
  int n = n0 + (l & 15); n = (n > maxCol) ? maxCol : n;
  int kh = (l >> 4) * 8;
  const __bf16* p = Bm + (size_t)n * (size_t)ld + (size_t)(k0 + kh);
  v8bf lo = *(const v8bf*)p;
  v8bf hi = *(const v8bf*)(p + 16);
  return __builtin_shufflevector(lo, hi, 0, 1, 2, 3, 4, 5, 6, 7,
                                 8, 9, 10, 11, 12, 13, 14, 15);
}

// ---- kernels ----

__global__ void cvt_f32_bf16(const float* __restrict__ s, __bf16* __restrict__ d, int n) {
  int i = blockIdx.x * blockDim.x + threadIdx.x;
  if (i < n) d[i] = (__bf16)s[i];
}

// x^T and diff^T share the input's (b,n,c) linear order -> pure elementwise, coalesced.
__global__ void prep_images(const float* __restrict__ bef, const float* __restrict__ aft,
                            __bf16* __restrict__ xT, __bf16* __restrict__ diffT) {
  size_t i = (size_t)blockIdx.x * blockDim.x + threadIdx.x;
  if (i >= nDiffT) return;
  float bv = bef[i];
  float dv = (aft[i] - bv) * (1.0f / (float)(kH - 1));
  diffT[i] = (__bf16)dv;
#pragma unroll
  for (int h = 0; h < kH; ++h)
    xT[(size_t)h * nDiffT + i] = (__bf16)(bv + (float)h * dv);
}

// Q[hb][n][o] = (Wq[h] @ x[hb])^T + bq
__global__ __launch_bounds__(32) void q_gemm(const __bf16* __restrict__ wq,
                                             const __bf16* __restrict__ xT,
                                             const float* __restrict__ bq,
                                             __bf16* __restrict__ qbf) {
  int hb = blockIdx.z, h = hb >> 4;
  int n0 = blockIdx.x * 16, o0 = blockIdx.y * 16;
  const __bf16* A = wq + (size_t)h * kC8 * kC;
  const __bf16* Bx = xT + (size_t)hb * kN * kC;
  v8f acc = {};
  for (int k0 = 0; k0 < kC; k0 += 32) {
    v16bf a = load_a16(A, kC, o0, k0, kC8 - 1);
    v16bf b = load_b16(Bx, kC, k0, n0, kN - 1);
    acc = WMMA_BF16(a, b, acc);
  }
  int l = (int)(threadIdx.x & 31u);
  int n = n0 + (l & 15), hi = l >> 4;
  if (n < kN) {
    v8bf pk;
#pragma unroll
    for (int j = 0; j < 8; ++j)
      pk[j] = (__bf16)(acc[j] + bq[h * kC8 + o0 + 8 * hi + j]);
    *(v8bf*)(qbf + (size_t)hb * kN * kC8 + (size_t)n * kC8 + o0 + 8 * hi) = pk;
  }
}

// K^T[hb][m][o] = h*(Wk[h] @ diff[b])^T + bk
__global__ __launch_bounds__(32) void k_gemm(const __bf16* __restrict__ wk,
                                             const __bf16* __restrict__ diffT,
                                             const float* __restrict__ bk,
                                             __bf16* __restrict__ ktb) {
  int hb = blockIdx.z, h = hb >> 4, b = hb & 15;
  int m0 = blockIdx.x * 16, o0 = blockIdx.y * 16;
  const __bf16* A = wk + (size_t)h * kC8 * kC;
  const __bf16* Bx = diffT + (size_t)b * kN * kC;
  v8f acc = {};
  for (int k0 = 0; k0 < kC; k0 += 32) {
    v16bf a = load_a16(A, kC, o0, k0, kC8 - 1);
    v16bf bb = load_b16(Bx, kC, k0, m0, kN - 1);
    acc = WMMA_BF16(a, bb, acc);
  }
  int l = (int)(threadIdx.x & 31u);
  int m = m0 + (l & 15), hi = l >> 4;
  float hs = (float)h;
  if (m < kN) {
    v8bf pk;
#pragma unroll
    for (int j = 0; j < 8; ++j)
      pk[j] = (__bf16)(hs * acc[j] + bk[h * kC8 + o0 + 8 * hi + j]);
    *(v8bf*)(ktb + (size_t)hb * kN * kC8 + (size_t)m * kC8 + o0 + 8 * hi) = pk;
  }
}

// energy[hb][n][m] = Q[hb][n][:] . K^T[hb][m][:]
__global__ __launch_bounds__(32) void e_gemm(const __bf16* __restrict__ qbf,
                                             const __bf16* __restrict__ ktb,
                                             float* __restrict__ energy) {
  int hb = blockIdx.z;
  int m0 = blockIdx.x * 16, n0 = blockIdx.y * 16;
  const __bf16* A = qbf + (size_t)hb * kN * kC8;
  const __bf16* Bm = ktb + (size_t)hb * kN * kC8;
  v8f acc = {};
  for (int k0 = 0; k0 < kC8; k0 += 32) {
    v16bf a = load_a16(A, kC8, n0, k0, kN - 1);
    v16bf b = load_b16(Bm, kC8, k0, m0, kN - 1);
    acc = WMMA_BF16(a, b, acc);
  }
  int l = (int)(threadIdx.x & 31u);
  int m = m0 + (l & 15), hi = l >> 4;
  if (m < kN) {
    float* dst = energy + (size_t)hb * kN * kN;
#pragma unroll
    for (int j = 0; j < 8; ++j) {
      int n = n0 + j + 8 * hi;
      if (n < kN) dst[(size_t)n * kN + m] = acc[j];
    }
  }
}

// per-row softmax; writes alphas (output layout) + padded row-major attn (bf16).
__global__ __launch_bounds__(256) void softmax_rows(const float* __restrict__ energy,
                                                    float* __restrict__ alphas,
                                                    __bf16* __restrict__ attn) {
  int nq = blockIdx.x, hb = blockIdx.y, h = hb >> 4, b = hb & 15;
  int t = threadIdx.x;
  const float* row = energy + ((size_t)hb * kN + nq) * kN;
  float v = (t < kN) ? row[t] : -3.4e38f;
  __shared__ float red[256];
  red[t] = v; __syncthreads();
#pragma unroll
  for (int s = 128; s > 0; s >>= 1) { if (t < s) red[t] = fmaxf(red[t], red[t + s]); __syncthreads(); }
  float mx = red[0]; __syncthreads();
  float e = (t < kN) ? __expf(v - mx) : 0.f;
  red[t] = e; __syncthreads();
#pragma unroll
  for (int s = 128; s > 0; s >>= 1) { if (t < s) red[t] += red[t + s]; __syncthreads(); }
  float inv = 1.f / red[0];
  float p = e * inv;
  if (t < kN)
    alphas[(((size_t)b * kN + nq) * kN + t) * kH + h] = p;
  if (t < kNP)  // zero-fill pad nk in [196,224)
    attn[((size_t)hb * kN + nq) * kNP + t] = (__bf16)((t < kN) ? p : 0.f);
}

// V[hb][c][n] = Wv[h] @ x[hb] + bv  (32x32 tile per wave, pad cols written as 0)
__global__ __launch_bounds__(32) void v_gemm(const __bf16* __restrict__ wv,
                                             const __bf16* __restrict__ xT,
                                             const float* __restrict__ bv,
                                             __bf16* __restrict__ vbf) {
  int hb = blockIdx.z, h = hb >> 4;
  int n0 = blockIdx.x * 32, c0 = blockIdx.y * 32;
  const __bf16* A = wv + (size_t)h * kC * kC;
  const __bf16* Bx = xT + (size_t)hb * kN * kC;
  v8f a00 = {}, a01 = {}, a10 = {}, a11 = {};
  for (int k0 = 0; k0 < kC; k0 += 32) {
    v16bf fa0 = load_a16(A, kC, c0, k0, kC - 1);
    v16bf fa1 = load_a16(A, kC, c0 + 16, k0, kC - 1);
    v16bf fb0 = load_b16(Bx, kC, k0, n0, kN - 1);
    v16bf fb1 = load_b16(Bx, kC, k0, n0 + 16, kN - 1);
    a00 = WMMA_BF16(fa0, fb0, a00);
    a01 = WMMA_BF16(fa0, fb1, a01);
    a10 = WMMA_BF16(fa1, fb0, a10);
    a11 = WMMA_BF16(fa1, fb1, a11);
  }
  int l = (int)(threadIdx.x & 31u);
  int cl = l & 15, hi = l >> 4;
  __bf16* dst = vbf + (size_t)hb * kC * kNP;
  v8f accs[4] = {a00, a01, a10, a11};
#pragma unroll
  for (int t = 0; t < 4; ++t) {
    int cb = c0 + (t >> 1) * 16, nb = n0 + (t & 1) * 16;
    int n = nb + cl;
#pragma unroll
    for (int j = 0; j < 8; ++j) {
      int c = cb + j + 8 * hi;
      float val = (n < kN) ? (accs[t][j] + bv[h * kC + c]) : 0.f;  // zero the pad
      dst[(size_t)c * kNP + n] = (__bf16)val;
    }
  }
}

// out^T[hb][m][c] = attn[hb] @ V[hb]^T ; feats = gamma*out + x  (32x32 per wave)
__global__ __launch_bounds__(32) void o_gemm(const __bf16* __restrict__ attn,
                                             const __bf16* __restrict__ vbf,
                                             const __bf16* __restrict__ xT,
                                             const float* __restrict__ gamma,
                                             __bf16* __restrict__ feats) {
  int hb = blockIdx.z, h = hb >> 4, b = hb & 15;
  int m0 = blockIdx.x * 32, c0 = blockIdx.y * 32;
  const __bf16* A = attn + (size_t)hb * kN * kNP;  // rows mq, k = nk (padded 0)
  const __bf16* Bm = vbf + (size_t)hb * kC * kNP;  // cols c,  k = nk (padded 0)
  v8f a00 = {}, a01 = {}, a10 = {}, a11 = {};
  for (int k0 = 0; k0 < kNP; k0 += 32) {  // 7 iters, no tail masking (pads are 0)
    v16bf fa0 = load_a16(A, kNP, m0, k0, kN - 1);
    v16bf fa1 = load_a16(A, kNP, m0 + 16, k0, kN - 1);
    v16bf fb0 = load_b16(Bm, kNP, k0, c0, kC - 1);
    v16bf fb1 = load_b16(Bm, kNP, k0, c0 + 16, kC - 1);
    a00 = WMMA_BF16(fa0, fb0, a00);
    a01 = WMMA_BF16(fa0, fb1, a01);
    a10 = WMMA_BF16(fa1, fb0, a10);
    a11 = WMMA_BF16(fa1, fb1, a11);
  }
  int l = (int)(threadIdx.x & 31u);
  int cl = l & 15, hi = l >> 4;
  float g = gamma[h];
  const __bf16* xs = xT + (size_t)hb * kN * kC;        // x[m][c]
  __bf16* dst = feats + (size_t)(b * kH + h) * kF;     // f = c*196 + m
  v8f accs[4] = {a00, a01, a10, a11};
#pragma unroll
  for (int t = 0; t < 4; ++t) {
    int mb = m0 + (t >> 1) * 16, cb = c0 + (t & 1) * 16;
    int c = cb + cl;
#pragma unroll
    for (int j = 0; j < 8; ++j) {
      int m = mb + j + 8 * hi;
      if (m < kN) {
        float res = g * accs[t][j] + (float)xs[(size_t)m * kC + c];
        dst[(size_t)c * kN + m] = (__bf16)res;
      }
    }
  }
}

__global__ void init_out(float* __restrict__ out, const float* __restrict__ bl) {
  int i = blockIdx.x * blockDim.x + threadIdx.x;
  if (i < kB * kH * kDM) out[i] = bl[i & (kDM - 1)];
}

// out[col][m] += Wl[m][:] . feats[col][:]  (split-K; Wl streamed exactly once)
__global__ __launch_bounds__(32) void final_gemm(const float* __restrict__ Wl,
                                                 const __bf16* __restrict__ feats,
                                                 float* __restrict__ out) {
  int m0 = blockIdx.x * 16;
  int kbase = blockIdx.y * kKChunk;
  v8f a0 = {}, a1 = {}, a2 = {}, a3 = {};
  for (int kk = 0; kk < kKChunk; kk += 32) {
    int k0 = kbase + kk;
    __builtin_prefetch(Wl + (size_t)(m0 + (int)(threadIdx.x & 15u)) * kF + k0 + 256, 0, 0);
    v16bf a  = load_a_f32(Wl, kF, m0, k0);
    v16bf b0 = load_b16(feats, kF, k0, 0, 63);
    v16bf b1 = load_b16(feats, kF, k0, 16, 63);
    v16bf b2 = load_b16(feats, kF, k0, 32, 63);
    v16bf b3 = load_b16(feats, kF, k0, 48, 63);
    a0 = WMMA_BF16(a, b0, a0);
    a1 = WMMA_BF16(a, b1, a1);
    a2 = WMMA_BF16(a, b2, a2);
    a3 = WMMA_BF16(a, b3, a3);
  }
  int l = (int)(threadIdx.x & 31u);
  int cl = l & 15, hi = l >> 4;
  v8f accs[4] = {a0, a1, a2, a3};
#pragma unroll
  for (int f = 0; f < 4; ++f) {
    int col = f * 16 + cl;
#pragma unroll
    for (int j = 0; j < 8; ++j) {
      int row = m0 + j + 8 * hi;
      __hip_atomic_fetch_add(&out[(size_t)col * kDM + row], accs[f][j],
                             __ATOMIC_RELAXED, __HIP_MEMORY_SCOPE_AGENT);
    }
  }
}

}  // namespace

extern "C" void kernel_launch(void* const* d_in, const int* in_sizes, int n_in,
                              void* d_out, int out_size, void* d_ws, size_t ws_size,
                              hipStream_t stream) {
  (void)in_sizes; (void)n_in; (void)out_size; (void)ws_size;
  const float* before = (const float*)d_in[0];
  const float* after  = (const float*)d_in[1];
  const float* Wq     = (const float*)d_in[2];
  const float* bq     = (const float*)d_in[3];
  const float* Wk     = (const float*)d_in[4];
  const float* bk     = (const float*)d_in[5];
  const float* Wv     = (const float*)d_in[6];
  const float* bv     = (const float*)d_in[7];
  const float* gamma  = (const float*)d_in[8];
  const float* Wl     = (const float*)d_in[9];
  const float* bl     = (const float*)d_in[10];

  char* ws = (char*)d_ws;
  __bf16* xT     = (__bf16*)(ws + oXT);
  __bf16* diffT  = (__bf16*)(ws + oDiffT);
  __bf16* wqbf   = (__bf16*)(ws + oWq);
  __bf16* wkbf   = (__bf16*)(ws + oWk);
  __bf16* wvbf   = (__bf16*)(ws + oWv);
  __bf16* qbf    = (__bf16*)(ws + oQ);
  __bf16* ktb    = (__bf16*)(ws + oKT);
  float*  energy = (float*) (ws + oE);
  __bf16* attn   = (__bf16*)(ws + oAT);
  __bf16* vbf    = (__bf16*)(ws + oV);
  __bf16* feats  = (__bf16*)(ws + oFeats);

  float* out0   = (float*)d_out;                 // (B, H, DM)
  float* alphas = out0 + (size_t)kB * kH * kDM;  // (B, 1, N, N, H)

  const int nT16 = (kN + 15) / 16;   // 13
  const int nT32 = (kNP) / 32;       // 7

  // 1) convert weights to bf16
  cvt_f32_bf16<<<(int)((nWq + 255) / 256), 256, 0, stream>>>(Wq, wqbf, (int)nWq);
  cvt_f32_bf16<<<(int)((nWk + 255) / 256), 256, 0, stream>>>(Wk, wkbf, (int)nWk);
  cvt_f32_bf16<<<(int)((nWv + 255) / 256), 256, 0, stream>>>(Wv, wvbf, (int)nWv);

  // 2) x-trajectory + diff (elementwise, coalesced; layout == input order)
  prep_images<<<(int)((nDiffT + 255) / 256), 256, 0, stream>>>(before, after, xT, diffT);

  // 3) Q/K projections
  q_gemm<<<dim3(nT16, kC8 / 16, kHB), 32, 0, stream>>>(wqbf, xT, bq, qbf);
  k_gemm<<<dim3(nT16, kC8 / 16, kHB), 32, 0, stream>>>(wkbf, diffT, bk, ktb);

  // 4) energy + softmax
  e_gemm<<<dim3(nT16, nT16, kHB), 32, 0, stream>>>(qbf, ktb, energy);
  softmax_rows<<<dim3(kN, kHB), 256, 0, stream>>>(energy, alphas, attn);

  // 5) V projection (dominant compute, 105 GFLOP; 32x32 tiles)
  v_gemm<<<dim3(nT32, kC / 32, kHB), 32, 0, stream>>>(wvbf, xT, bv, vbf);

  // 6) attn @ V^T + gamma residual -> feats (32x32 tiles)
  o_gemm<<<dim3(nT32, kC / 32, kHB), 32, 0, stream>>>(attn, vbf, xT, gamma, feats);

  // 7) final linear: bias init, then split-K WMMA GEMM with f32 atomics
  init_out<<<(kB * kH * kDM + 255) / 256, 256, 0, stream>>>(out0, bl);
  final_gemm<<<dim3(kDM / 16, kKSplit), 32, 0, stream>>>(Wl, feats, out0);
}
